// TrafficTransformerEncoder_3762391351339
// MI455X (gfx1250) — matrix operand order repaired and checked by reading, not compiled, not run on hardware
//
#include <hip/hip_runtime.h>

// ---------------- types & helpers ----------------
typedef unsigned short u16;
typedef __attribute__((ext_vector_type(16))) __bf16 v16bf;
typedef __attribute__((ext_vector_type(8)))  float  v8f;
typedef __attribute__((ext_vector_type(4)))  unsigned int uint4v;

union FragU { v16bf v; uint4v q[2]; u16 u[16]; };

static constexpr int Bb = 16, Ns = 64, Tt = 168, Ff = 8, Dd = 248;
static constexpr int Ee = 256, Hh = 8, DH = 32, FFd = 1024;
static constexpr int ROWS = Bb * Ns * Tt;   // 172032
static constexpr int BT   = Ns * Tt;        // 10752

__device__ __forceinline__ u16 f2bf(float f) {
    union { float f; unsigned u; } x; x.f = f;
    unsigned r = x.u + 0x7FFFu + ((x.u >> 16) & 1u);
    return (u16)(r >> 16);
}
__device__ __forceinline__ float bf2f(u16 h) {
    union { unsigned u; float f; } x; x.u = ((unsigned)h) << 16;
    return x.f;
}
__device__ __forceinline__ v8f vzero8() {
    v8f z;
#pragma unroll
    for (int i = 0; i < 8; i++) z[i] = 0.0f;
    return z;
}
__device__ __forceinline__ v8f wmma_bf(v16bf a, v16bf b, v8f c) {
    return __builtin_amdgcn_wmma_f32_16x16x32_bf16(false, a, false, b, (short)0, c, false, false);
}
// A fragment (16x32 bf16, row-major source, lda in elements).
// lane L holds row M = m0 + (L&15); K = k0 + 16*(L>=16) + i (16 contiguous). Two 16B loads.
__device__ __forceinline__ v16bf load_a_frag(const u16* A, int lda, int m0, int k0, int lane) {
    const u16* p = A + (size_t)(m0 + (lane & 15)) * (size_t)lda + (size_t)(k0 + ((lane >> 4) << 4));
    FragU f;
    f.q[0] = *(const uint4v*)(p);
    f.q[1] = *(const uint4v*)(p + 8);
    return f.v;
}
// B fragment from TRANSPOSED weights Wt[N][K] (row-major, ldk).
// lane L holds column N = n0 + (L&15), K = k0 + 16*(L>=16) + i (16 contiguous).
__device__ __forceinline__ v16bf load_bt_frag(const u16* WT, int ldk, int n0, int k0, int lane) {
    const u16* p = WT + (size_t)(n0 + (lane & 15)) * (size_t)ldk + (size_t)(k0 + ((lane >> 4) << 4));
    FragU f;
    f.q[0] = *(const uint4v*)(p);
    f.q[1] = *(const uint4v*)(p + 8);
    return f.v;
}

// ---------------- prep kernels ----------------
// dst[n*Kp + k] = src[k*N + n] (bf16), zero padded
__global__ void k_transpose_pad(const float* __restrict__ src, u16* __restrict__ dst,
                                int K, int N, int Kp, int Np) {
    int idx = blockIdx.x * 256 + threadIdx.x;
    if (idx >= Kp * Np) return;
    int n = idx / Kp, k = idx % Kp;
    float v = (k < K && n < N) ? src[(size_t)k * N + n] : 0.0f;
    dst[idx] = f2bf(v);
}

// normalized adjacency: a = (w_rownorm + I) * 0.5  (row sums of w+I are exactly 2)
__global__ void k_adj(const float* __restrict__ locs, u16* __restrict__ a_bf) {
    int b = blockIdx.x, m = threadIdx.x;
    const float* L = locs + b * Ns * 2;
    float mx = L[m * 2 + 0], my = L[m * 2 + 1];
    float wsum = 0.0f;
    for (int n = 0; n < Ns; n++) {
        float dx = mx - L[n * 2 + 0], dy = my - L[n * 2 + 1];
        if (n != m) wsum += rsqrtf(dx * dx + dy * dy + 1e-12f);
    }
    float rw = 0.5f / wsum;
    for (int n = 0; n < Ns; n++) {
        float dx = mx - L[n * 2 + 0], dy = my - L[n * 2 + 1];
        float v = (n == m) ? 0.5f : rsqrtf(dx * dx + dy * dy + 1e-12f) * rw;
        a_bf[b * Ns * Ns + m * Ns + n] = f2bf(v);
    }
}

// h[row][0:256] = x[row][0:8] @ gcn_w (K=8), bf16, cols 248..255 zero
__global__ void k_gcn_h(const float* __restrict__ x, const float* __restrict__ gcn_w,
                        u16* __restrict__ h) {
    int r = blockIdx.x, d = threadIdx.x;
    float acc = 0.0f;
    if (d < Dd) {
#pragma unroll
        for (int f = 0; f < Ff; f++) acc += x[(size_t)r * Ff + f] * gcn_w[f * Dd + d];
    }
    h[(size_t)r * Ee + d] = f2bf(d < Dd ? acc : 0.0f);
}

// tok cols 0..7 = x + pos ; qkin = tok + per
__global__ void k_tokx(const float* __restrict__ x, const float* __restrict__ pos_em,
                       const float* __restrict__ per_em, u16* __restrict__ tok_bf,
                       u16* __restrict__ qkin) {
    int idx = blockIdx.x * 256 + threadIdx.x;   // ROWS*8
    int r = idx >> 3, e = idx & 7;
    int bb = r / BT, tt = r % Tt;
    size_t pidx = ((size_t)(bb * Tt) + tt) * Ee + e;
    float v = x[idx] + pos_em[pidx];
    tok_bf[(size_t)r * Ee + e] = f2bf(v);
    qkin[(size_t)r * Ee + e]   = f2bf(v + per_em[pidx]);
}

// ---------------- WMMA kernels ----------------
// g1[b,:,t,:] = relu(a[b] @ h[b,:,t,:] + gcn_b); M=64 stations, K=64, N=256
__global__ void __launch_bounds__(256) k_aggregate(const u16* __restrict__ h,
                                                   const u16* __restrict__ a_bf,
                                                   const float* __restrict__ gcn_b,
                                                   u16* __restrict__ g1) {
    __shared__ __align__(16) u16 ht[Ee * Ns];   // [dim][station], 32KB
    int b = blockIdx.x / Tt, t = blockIdx.x % Tt;
    int tid = threadIdx.x, lane = tid & 31, w = tid >> 5;
    for (int s = 0; s < Ns; s++)
        ht[tid * Ns + s] = h[((size_t)((b * Ns + s) * Tt) + t) * Ee + tid];
    __syncthreads();

    int mt = (w & 3) * 16;          // station-out tile
    int n0 = (w >> 2) * 128;        // dim base (each wave: 8 n-tiles)
    v8f acc[8];
#pragma unroll
    for (int j = 0; j < 8; j++) acc[j] = vzero8();
    const u16* Ab = a_bf + b * Ns * Ns;
#pragma unroll
    for (int k0 = 0; k0 < 64; k0 += 32) {
        v16bf af = load_a_frag(Ab, Ns, mt, k0, lane);
#pragma unroll
        for (int j = 0; j < 8; j++)
            acc[j] = wmma_bf(af, load_bt_frag(ht, Ns, n0 + 16 * j, k0, lane), acc[j]);
    }
    int hi8 = (lane >> 4) << 3, cn = lane & 15;
#pragma unroll
    for (int j = 0; j < 8; j++) {
        int c = n0 + 16 * j + cn;
        float bias = (c < Dd) ? gcn_b[c] : 0.0f;
#pragma unroll
        for (int r = 0; r < 8; r++) {
            int m = mt + r + hi8;
            float v = acc[j][r] + bias;
            v = v > 0.0f ? v : 0.0f;
            g1[((size_t)((b * Ns + m) * Tt) + t) * Ee + c] = f2bf(v);
        }
    }
}

// fc1 + concat + embeddings: tok[.,8+d] = g1@fc1 + b + pos ; qkin = tok + per
__global__ void __launch_bounds__(256) k_fc1(const u16* __restrict__ g1, const u16* __restrict__ wT,
                                             const float* __restrict__ fc1_b,
                                             const float* __restrict__ pos_em,
                                             const float* __restrict__ per_em,
                                             u16* __restrict__ tok_bf, u16* __restrict__ qkin) {
    int tid = threadIdx.x, lane = tid & 31, w = tid >> 5;
    int m0 = blockIdx.x * 128 + w * 16;
    v8f acc[16];
#pragma unroll
    for (int j = 0; j < 16; j++) acc[j] = vzero8();
    for (int k0 = 0; k0 < Ee; k0 += 32) {
        v16bf af = load_a_frag(g1, Ee, m0, k0, lane);
#pragma unroll
        for (int j = 0; j < 16; j++)
            acc[j] = wmma_bf(af, load_bt_frag(wT, Ee, 16 * j, k0, lane), acc[j]);
    }
    int hi8 = (lane >> 4) << 3, cn = lane & 15;
#pragma unroll
    for (int j = 0; j < 16; j++) {
        int c = 16 * j + cn;
        if (c < Dd) {
            int e = Ff + c;
            float bias = fc1_b[c];
#pragma unroll
            for (int r = 0; r < 8; r++) {
                int row = m0 + r + hi8;
                int bb = row / BT, tt = row % Tt;
                size_t pidx = ((size_t)(bb * Tt) + tt) * Ee + e;
                float tv = acc[j][r] + bias + pos_em[pidx];
                tok_bf[(size_t)row * Ee + e] = f2bf(tv);
                qkin[(size_t)row * Ee + e]   = f2bf(tv + per_em[pidx]);
            }
        }
    }
}

// generic 256x256 projection with bias -> bf16 (Q/K/V)
__global__ void __launch_bounds__(256) k_gemm_bias(const u16* __restrict__ A,
                                                   const u16* __restrict__ wT,
                                                   const float* __restrict__ bias,
                                                   u16* __restrict__ outb) {
    int tid = threadIdx.x, lane = tid & 31, w = tid >> 5;
    int m0 = blockIdx.x * 128 + w * 16;
    v8f acc[16];
#pragma unroll
    for (int j = 0; j < 16; j++) acc[j] = vzero8();
    for (int k0 = 0; k0 < Ee; k0 += 32) {
        v16bf af = load_a_frag(A, Ee, m0, k0, lane);
#pragma unroll
        for (int j = 0; j < 16; j++)
            acc[j] = wmma_bf(af, load_bt_frag(wT, Ee, 16 * j, k0, lane), acc[j]);
    }
    int hi8 = (lane >> 4) << 3, cn = lane & 15;
#pragma unroll
    for (int j = 0; j < 16; j++) {
        int c = 16 * j + cn;
#pragma unroll
        for (int r = 0; r < 8; r++) {
            int row = m0 + r + hi8;
            outb[(size_t)row * Ee + c] = f2bf(acc[j][r] + bias[c]);
        }
    }
}

// attention: one wave per (bn, head, 16-row tile). S in LDS f32, softmax, P@V via WMMA.
__global__ void __launch_bounds__(64) k_attn(const u16* __restrict__ qg, const u16* __restrict__ kg,
                                             const u16* __restrict__ vg, u16* __restrict__ og) {
    __shared__ __align__(16) float ss[2][16 * 176];
    __shared__ __align__(16) u16   pp[2][16 * 192];
    int tid = threadIdx.x, lane = tid & 31, w = tid >> 5;
    int bn = blockIdx.x >> 3, hd = blockIdx.x & 7;
    int tile = blockIdx.y * 2 + w;
    if (tile >= 11) return;   // whole-wave uniform

    float* s = ss[w];
    u16*   p = pp[w];
    int m0 = tile * 16;
    size_t rowbase = (size_t)bn * Tt;

    // ---- S = Q @ K^T (K-dim = 32, one step) ----
    const u16* Aq = qg + rowbase * Ee + hd * DH;
    v16bf af = load_a_frag(Aq, Ee, m0, 0, lane);
    v8f sacc[11];
#pragma unroll
    for (int j = 0; j < 11; j++) {
        int tc = j * 16 + (lane & 15);
        int kb = (lane >> 4) << 4;
        FragU f;
        if (tc < Tt) {
            const u16* pk = kg + (rowbase + tc) * Ee + hd * DH + kb;
            f.q[0] = *(const uint4v*)(pk);
            f.q[1] = *(const uint4v*)(pk + 8);
        } else {
#pragma unroll
            for (int i = 0; i < 16; i++) f.u[i] = 0;
        }
        sacc[j] = wmma_bf(af, f.v, vzero8());
    }
    const float scale = 0.17677669529663687f;   // 1/sqrt(32)
    int hi8 = (lane >> 4) << 3, cn = lane & 15;
#pragma unroll
    for (int j = 0; j < 11; j++)
#pragma unroll
        for (int r = 0; r < 8; r++)
            s[(r + hi8) * 176 + j * 16 + cn] = sacc[j][r] * scale;

    // ---- softmax: lanes 0..15, one row each (LDS ops are in-order within a wave) ----
    if (lane < 16) {
        int tg = m0 + lane;
        if (tg < Tt) {
            float mx = -1e30f;
            for (int c = 0; c < Tt; c++) mx = fmaxf(mx, s[lane * 176 + c]);
            float sum = 0.0f;
            for (int c = 0; c < Tt; c++) {
                float e = __expf(s[lane * 176 + c] - mx);
                s[lane * 176 + c] = e;
                sum += e;
            }
            float rs = 1.0f / sum;
            for (int c = 0; c < 192; c++)
                p[lane * 192 + c] = (c < Tt) ? f2bf(s[lane * 176 + c] * rs) : (u16)0;
        } else {
            for (int c = 0; c < 192; c++) p[lane * 192 + c] = 0;
        }
    }

    // ---- O = P @ V  (K padded to 192; padded P cols are zero) ----
    v8f oacc[2];
    oacc[0] = vzero8(); oacc[1] = vzero8();
    for (int k0 = 0; k0 < 192; k0 += 32) {
        v16bf pf = load_a_frag(p, 192, 0, k0, lane);
#pragma unroll
        for (int j = 0; j < 2; j++) {
            int dcol = j * 16 + (lane & 15);
            int kb = k0 + ((lane >> 4) << 4);
            FragU f;
#pragma unroll
            for (int i = 0; i < 16; i++) {
                int kk = kb + i;
                f.u[i] = (kk < Tt) ? vg[(rowbase + kk) * Ee + hd * DH + dcol] : (u16)0;
            }
            oacc[j] = wmma_bf(pf, f.v, oacc[j]);
        }
    }
#pragma unroll
    for (int j = 0; j < 2; j++) {
        int c = hd * DH + j * 16 + cn;
#pragma unroll
        for (int r = 0; r < 8; r++) {
            int tg = m0 + r + hi8;
            if (tg < Tt) og[(rowbase + tg) * Ee + c] = f2bf(oacc[j][r]);
        }
    }
}

// Wo projection + residual + LayerNorm1 -> h1 bf16. BM=32 rows per block.
__global__ void __launch_bounds__(256) k_wo_ln1(const u16* __restrict__ o, const u16* __restrict__ wT,
                                                const float* __restrict__ bo,
                                                const u16* __restrict__ tok_bf,
                                                const float* __restrict__ g, const float* __restrict__ be,
                                                u16* __restrict__ h1) {
    __shared__ __align__(16) float s[32 * 256];
    __shared__ float lm[32], lr[32];
    int tid = threadIdx.x, lane = tid & 31, w = tid >> 5;
    int mb = blockIdx.x * 32;
    int m0 = mb + (w & 1) * 16;
    int n0 = (w >> 1) * 64;
    v8f acc[4];
#pragma unroll
    for (int j = 0; j < 4; j++) acc[j] = vzero8();
    for (int k0 = 0; k0 < Ee; k0 += 32) {
        v16bf af = load_a_frag(o, Ee, m0, k0, lane);
#pragma unroll
        for (int j = 0; j < 4; j++)
            acc[j] = wmma_bf(af, load_bt_frag(wT, Ee, n0 + 16 * j, k0, lane), acc[j]);
    }
    int hi8 = (lane >> 4) << 3, cn = lane & 15;
#pragma unroll
    for (int j = 0; j < 4; j++) {
        int c = n0 + 16 * j + cn;
#pragma unroll
        for (int r = 0; r < 8; r++) {
            int ml = (w & 1) * 16 + r + hi8;
            int row = mb + ml;
            s[ml * 256 + c] = acc[j][r] + bo[c] + bf2f(tok_bf[(size_t)row * Ee + c]);
        }
    }
    __syncthreads();
    if (tid < 32) {
        float su = 0.0f;
        for (int c = 0; c < 256; c++) su += s[tid * 256 + c];
        float mean = su * (1.0f / 256.0f);
        float vs = 0.0f;
        for (int c = 0; c < 256; c++) { float d = s[tid * 256 + c] - mean; vs += d * d; }
        lm[tid] = mean;
        lr[tid] = rsqrtf(vs * (1.0f / 256.0f) + 1e-5f);
    }
    __syncthreads();
    int rr = tid >> 3, c0 = (tid & 7) * 32;
    float mean = lm[rr], rstd = lr[rr];
    int row = mb + rr;
    for (int c = c0; c < c0 + 32; c++)
        h1[(size_t)row * Ee + c] = f2bf((s[rr * 256 + c] - mean) * rstd * g[c] + be[c]);
}

// fused FFN: relu(h1@W1+b1) in LDS (bf16, 64KB), then @W2+b2 + residual + LayerNorm2 -> fp32 out
__global__ void __launch_bounds__(256) k_ffn(const u16* __restrict__ h1, const u16* __restrict__ w1T,
                                             const float* __restrict__ b1,
                                             const u16* __restrict__ w2T, const float* __restrict__ b2,
                                             const float* __restrict__ g, const float* __restrict__ be,
                                             float* __restrict__ out) {
    __shared__ __align__(16) u16 smem_u[32 * 1024];   // stage1 out; overlaid later by f32 s2 + stats
    int tid = threadIdx.x, lane = tid & 31, w = tid >> 5;
    int mb = blockIdx.x * 32;
    int m0 = mb + (w & 1) * 16;
    int hi8 = (lane >> 4) << 3, cn = lane & 15;

    // ---- stage1: ffa = relu(h1 @ W1 + b1), [32 x 1024] bf16 in LDS ----
    {
        int n0 = (w >> 1) * 256;
        v8f acc[16];
#pragma unroll
        for (int j = 0; j < 16; j++) acc[j] = vzero8();
        for (int k0 = 0; k0 < Ee; k0 += 32) {
            v16bf af = load_a_frag(h1, Ee, m0, k0, lane);
#pragma unroll
            for (int j = 0; j < 16; j++)
                acc[j] = wmma_bf(af, load_bt_frag(w1T, Ee, n0 + 16 * j, k0, lane), acc[j]);
        }
#pragma unroll
        for (int j = 0; j < 16; j++) {
            int c = n0 + 16 * j + cn;
            float bias = b1[c];
#pragma unroll
            for (int r = 0; r < 8; r++) {
                int ml = (w & 1) * 16 + r + hi8;
                float v = acc[j][r] + bias;
                smem_u[ml * 1024 + c] = f2bf(v > 0.0f ? v : 0.0f);
            }
        }
    }
    __syncthreads();

    // ---- stage2: acc2 = ffa @ W2 (K=1024, A from LDS) ----
    v8f acc2[4];
#pragma unroll
    for (int j = 0; j < 4; j++) acc2[j] = vzero8();
    int n0 = (w >> 1) * 64;
    for (int k0 = 0; k0 < FFd; k0 += 32) {
        v16bf af = load_a_frag(smem_u, FFd, (w & 1) * 16, k0, lane);
#pragma unroll
        for (int j = 0; j < 4; j++)
            acc2[j] = wmma_bf(af, load_bt_frag(w2T, FFd, n0 + 16 * j, k0, lane), acc2[j]);
    }
    __syncthreads();   // all LDS reads of ffa consumed by wmma above; safe to overlay

    float* s2 = (float*)smem_u;                 // 32*256 f32 = 32KB
    float* lm = (float*)smem_u + 32 * 256;      // stats live past the s2 region
    float* lr = lm + 32;
#pragma unroll
    for (int j = 0; j < 4; j++) {
        int c = n0 + 16 * j + cn;
#pragma unroll
        for (int r = 0; r < 8; r++) {
            int ml = (w & 1) * 16 + r + hi8;
            int row = mb + ml;
            s2[ml * 256 + c] = acc2[j][r] + b2[c] + bf2f(h1[(size_t)row * Ee + c]);
        }
    }
    __syncthreads();
    if (tid < 32) {
        float su = 0.0f;
        for (int c = 0; c < 256; c++) su += s2[tid * 256 + c];
        float mean = su * (1.0f / 256.0f);
        float vs = 0.0f;
        for (int c = 0; c < 256; c++) { float d = s2[tid * 256 + c] - mean; vs += d * d; }
        lm[tid] = mean;
        lr[tid] = rsqrtf(vs * (1.0f / 256.0f) + 1e-5f);
    }
    __syncthreads();
    int rr = tid >> 3, c0 = (tid & 7) * 32;
    float mean = lm[rr], rstd = lr[rr];
    int row = mb + rr;
    for (int c = c0; c < c0 + 32; c++)
        out[(size_t)row * Ee + c] = (s2[rr * 256 + c] - mean) * rstd * g[c] + be[c];
}

// ---------------- host launcher ----------------
extern "C" void kernel_launch(void* const* d_in, const int* in_sizes, int n_in,
                              void* d_out, int out_size, void* d_ws, size_t ws_size,
                              hipStream_t stream) {
    (void)in_sizes; (void)n_in; (void)out_size; (void)ws_size;
    const float* x      = (const float*)d_in[0];
    const float* locs   = (const float*)d_in[1];
    const float* pos_em = (const float*)d_in[2];
    const float* per_em = (const float*)d_in[3];
    const float* gcn_w  = (const float*)d_in[4];
    const float* gcn_b  = (const float*)d_in[5];
    const float* fc1_w  = (const float*)d_in[6];
    const float* fc1_b  = (const float*)d_in[7];
    const float* wq     = (const float*)d_in[8];
    const float* bq     = (const float*)d_in[9];
    const float* wk     = (const float*)d_in[10];
    const float* bk     = (const float*)d_in[11];
    const float* wv     = (const float*)d_in[12];
    const float* bv     = (const float*)d_in[13];
    const float* wo     = (const float*)d_in[14];
    const float* bo     = (const float*)d_in[15];
    const float* ln1_g  = (const float*)d_in[16];
    const float* ln1_b  = (const float*)d_in[17];
    const float* ln2_g  = (const float*)d_in[18];
    const float* ln2_b  = (const float*)d_in[19];
    const float* ff1_w  = (const float*)d_in[20];
    const float* ff1_b  = (const float*)d_in[21];
    const float* ff2_w  = (const float*)d_in[22];
    const float* ff2_b  = (const float*)d_in[23];
    float* out = (float*)d_out;

    char* ws = (char*)d_ws;
    u16* a_bf  = (u16*)(ws + (0 << 10));
    u16* fc1wT = (u16*)(ws + (128 << 10));
    u16* wqT   = (u16*)(ws + (256 << 10));
    u16* wkT   = (u16*)(ws + (384 << 10));
    u16* wvT   = (u16*)(ws + (512 << 10));
    u16* woT   = (u16*)(ws + (640 << 10));
    u16* ff1wT = (u16*)(ws + (768 << 10));
    u16* ff2wT = (u16*)(ws + (1280 << 10));
    size_t SZ = (size_t)ROWS * Ee * sizeof(u16);   // 88080384
    char* bufs = ws + (2u << 20);
    u16* B1 = (u16*)(bufs + 0 * SZ);  // h -> later Q
    u16* B2 = (u16*)(bufs + 1 * SZ);  // g1 -> later K
    u16* B3 = (u16*)(bufs + 2 * SZ);  // tok (bf16)
    u16* B4 = (u16*)(bufs + 3 * SZ);  // qkin -> later O
    u16* B5 = (u16*)(bufs + 4 * SZ);  // V
    u16* B6 = (u16*)(bufs + 5 * SZ);  // h1 (bf16)

    // weights -> transposed/padded bf16
    k_transpose_pad<<<256, 256, 0, stream>>>(fc1_w, fc1wT, Dd, Dd, Ee, Ee);
    k_transpose_pad<<<256, 256, 0, stream>>>(wq, wqT, Ee, Ee, Ee, Ee);
    k_transpose_pad<<<256, 256, 0, stream>>>(wk, wkT, Ee, Ee, Ee, Ee);
    k_transpose_pad<<<256, 256, 0, stream>>>(wv, wvT, Ee, Ee, Ee, Ee);
    k_transpose_pad<<<256, 256, 0, stream>>>(wo, woT, Ee, Ee, Ee, Ee);
    k_transpose_pad<<<1024, 256, 0, stream>>>(ff1_w, ff1wT, Ee, FFd, Ee, FFd);
    k_transpose_pad<<<1024, 256, 0, stream>>>(ff2_w, ff2wT, FFd, Ee, FFd, Ee);

    k_adj<<<Bb, Ns, 0, stream>>>(locs, a_bf);
    k_gcn_h<<<ROWS, 256, 0, stream>>>(x, gcn_w, B1);
    k_aggregate<<<Bb * Tt, 256, 0, stream>>>(B1, a_bf, gcn_b, B2);
    k_fc1<<<ROWS / 128, 256, 0, stream>>>(B2, fc1wT, fc1_b, pos_em, per_em, B3, B4);
    k_tokx<<<ROWS * 8 / 256, 256, 0, stream>>>(x, pos_em, per_em, B3, B4);

    k_gemm_bias<<<ROWS / 128, 256, 0, stream>>>(B3, wvT, bv, B5);   // V from tok
    k_gemm_bias<<<ROWS / 128, 256, 0, stream>>>(B4, wqT, bq, B1);   // Q from qkin
    k_gemm_bias<<<ROWS / 128, 256, 0, stream>>>(B4, wkT, bk, B2);   // K from qkin

    k_attn<<<dim3(Bb * Ns * Hh, 6), 64, 0, stream>>>(B1, B2, B5, B4);  // O -> B4

    k_wo_ln1<<<ROWS / 32, 256, 0, stream>>>(B4, woT, bo, B3, ln1_g, ln1_b, B6);
    k_ffn<<<ROWS / 32, 256, 0, stream>>>(B6, ff1wT, ff1_b, ff2wT, ff2_b, ln2_g, ln2_b, out);
}
